// CompositeMHA_41205916238369
// MI455X (gfx1250) — matrix operand order repaired and checked by reading, not compiled, not run on hardware
//
#include <hip/hip_runtime.h>

// ---- problem constants (B=4, S=1024, E=2048, H=16, D=128) ----
#define BB 4
#define SS 1024
#define EE 2048
#define HH 16
#define DD 128

typedef __bf16 v16bf __attribute__((ext_vector_type(16)));
typedef float v8f __attribute__((ext_vector_type(8)));
typedef unsigned int u32x4 __attribute__((ext_vector_type(4)));
typedef int i32x4 __attribute__((ext_vector_type(4)));
typedef float f32x4 __attribute__((ext_vector_type(4)));

union FragU { v16bf bf; u32x4 u[2]; };

#if defined(__has_builtin)
#if __has_builtin(__builtin_amdgcn_cvt_pk_bf16_f32)
#define HAVE_CVT_PK_BF16 1
#endif
#if __has_builtin(__builtin_amdgcn_global_load_async_to_lds_b128) && \
    __has_builtin(__builtin_amdgcn_s_wait_asynccnt)
#define HAVE_ASYNC_LDS 1
#endif
#endif
#ifndef HAVE_CVT_PK_BF16
#define HAVE_CVT_PK_BF16 0
#endif
#ifndef HAVE_ASYNC_LDS
#define HAVE_ASYNC_LDS 0
#endif

__device__ __forceinline__ unsigned short f32_to_bf16(float f) {
  unsigned int u = __builtin_bit_cast(unsigned int, f);
  unsigned int r = u + 0x7FFFu + ((u >> 16) & 1u);  // round-to-nearest-even
  return (unsigned short)(r >> 16);
}

// pack two f32 into {lo, hi} bf16 pair
__device__ __forceinline__ unsigned int pack2_bf16(float x, float y) {
#if HAVE_CVT_PK_BF16
  auto r = __builtin_amdgcn_cvt_pk_bf16_f32(x, y);  // v_cvt_pk_bf16_f32
  return __builtin_bit_cast(unsigned int, r);
#else
  return (unsigned int)f32_to_bf16(x) | ((unsigned int)f32_to_bf16(y) << 16);
#endif
}

// A-matrix fragment (16x32 MxK, bf16). ISA layout: lane<16 -> row M=lane,
// K = {0..7, 16..23}; lane>=16 -> row M=lane-16, K = {8..15, 24..31}.
__device__ __forceinline__ v16bf load_fragA(const unsigned short* base, int pitch, int row0) {
  const int l = (int)(threadIdx.x & 31);
  const unsigned short* p = base + (row0 + (l & 15)) * pitch + ((l >> 4) << 3);
  FragU f;
  f.u[0] = *(const u32x4*)p;         // K = k0 .. k0+7
  f.u[1] = *(const u32x4*)(p + 16);  // K = k0+16 .. k0+23
  return f.bf;
}

// B-matrix fragment (32x16 KxN, bf16) from an N-major [n][k] buffer.
// ISA layout: lane<16 -> col N=lane, K = {0..7, 16..23}; lane>=16 -> col N=lane-16, K = {8..15, 24..31}.
__device__ __forceinline__ v16bf load_fragB(const unsigned short* base, int pitch, int col0) {
  const int l = (int)(threadIdx.x & 31);
  const unsigned short* p = base + (col0 + (l & 15)) * pitch + ((l >> 4) << 3);
  FragU f;
  f.u[0] = *(const u32x4*)p;         // K = k0 .. k0+7
  f.u[1] = *(const u32x4*)(p + 16);  // K = k0+16 .. k0+23
  return f.bf;
}

__device__ __forceinline__ v8f wmma_bf16(v16bf a, v16bf b, v8f c) {
  return __builtin_amdgcn_wmma_f32_16x16x32_bf16(false, a, false, b, (short)0, c, false, false);
}

// store 16 staged f32 (4x f32x4 regs) as 16 bf16 into LDS
__device__ __forceinline__ void store16_bf16(const f32x4* v, unsigned short* dst) {
  unsigned int* d = (unsigned int*)dst;
#pragma unroll
  for (int i = 0; i < 4; ++i) {
    d[2 * i + 0] = pack2_bf16(v[i].x, v[i].y);
    d[2 * i + 1] = pack2_bf16(v[i].z, v[i].w);
  }
}

// ---------------------------------------------------------------------------
// Generic C[M,N] = A[M,K] * W[N,K]^T + bias[N], register-pipelined staging.
// MODE 0: scatter to Q[B,H,S,D] / K[B,H,S,D] / V^T[B,H,D,S] (bf16)
// MODE 1: write f32 Out[M,N]
// ---------------------------------------------------------------------------
template <int A_BF16, int MODE>
__global__ __launch_bounds__(256) void gemm_xwt_kernel(
    const void* __restrict__ Aptr,
    const float* __restrict__ W, const float* __restrict__ bias,
    int M, int N, int K,
    unsigned short* __restrict__ Qb, unsigned short* __restrict__ Kb,
    unsigned short* __restrict__ Vtb, float* __restrict__ Out)
{
  __shared__ unsigned short lA[128 * 32];
  __shared__ unsigned short lW[128 * 32];
  const int tid = (int)threadIdx.x;
  const int lane = tid & 31, wave = tid >> 5;
  const int wm = wave & 3, wn = wave >> 2;          // 4 waves along M, 2 along N
  const int m0 = (int)blockIdx.x * 128, n0 = (int)blockIdx.y * 128;

  const v8f vzero = {0.f, 0.f, 0.f, 0.f, 0.f, 0.f, 0.f, 0.f};
  v8f acc[2][4];
#pragma unroll
  for (int mi = 0; mi < 2; ++mi)
#pragma unroll
    for (int ni = 0; ni < 4; ++ni) acc[mi][ni] = vzero;

  const int srow = tid >> 1;            // 0..127
  const int scol = (tid & 1) * 16;      // 0 or 16

  const float* arow_f = (const float*)Aptr + (size_t)(m0 + srow) * K + scol;
  const unsigned short* arow_h = (const unsigned short*)Aptr + (size_t)(m0 + srow) * K + scol;
  const float* wrow = W + (size_t)(n0 + srow) * K + scol;

  // --- register double-buffer: prologue load of k0 = 0 ---
  f32x4 areg_f[4];
  u32x4 areg_h[2];
  f32x4 wreg[4];
  if (A_BF16) {
    areg_h[0] = *(const u32x4*)(arow_h);
    areg_h[1] = *(const u32x4*)(arow_h + 8);
  } else {
#pragma unroll
    for (int i = 0; i < 4; ++i) areg_f[i] = *(const f32x4*)(arow_f + i * 4);
  }
#pragma unroll
  for (int i = 0; i < 4; ++i) wreg[i] = *(const f32x4*)(wrow + i * 4);

  for (int k0 = 0; k0 < K; k0 += 32) {
    // commit staged registers to LDS (convert f32 -> bf16 on the way)
    if (A_BF16) {
      *(u32x4*)&lA[srow * 32 + scol] = areg_h[0];
      *(u32x4*)&lA[srow * 32 + scol + 8] = areg_h[1];
    } else {
      store16_bf16(areg_f, &lA[srow * 32 + scol]);
    }
    store16_bf16(wreg, &lW[srow * 32 + scol]);
    __syncthreads();

    // preload next k-step while WMMAs run (overlaps global latency)
    if (k0 + 32 < K) {
      const int kn = k0 + 32;
      if (A_BF16) {
        areg_h[0] = *(const u32x4*)(arow_h + kn);
        areg_h[1] = *(const u32x4*)(arow_h + kn + 8);
      } else {
#pragma unroll
        for (int i = 0; i < 4; ++i) areg_f[i] = *(const f32x4*)(arow_f + kn + i * 4);
      }
#pragma unroll
      for (int i = 0; i < 4; ++i) wreg[i] = *(const f32x4*)(wrow + kn + i * 4);
    }

    v16bf bfrag[4];
#pragma unroll
    for (int ni = 0; ni < 4; ++ni) bfrag[ni] = load_fragB(lW, 32, wn * 64 + ni * 16);
#pragma unroll
    for (int mi = 0; mi < 2; ++mi) {
      v16bf afrag = load_fragA(lA, 32, wm * 32 + mi * 16);
#pragma unroll
      for (int ni = 0; ni < 4; ++ni)
        acc[mi][ni] = wmma_bf16(afrag, bfrag[ni], acc[mi][ni]);
    }
    __syncthreads();
  }

  // Epilogue. C-layout: reg r of lane<16 -> row r, lane>=16 -> row r+8; col = lane&15.
  const int half = lane >> 4, lc = lane & 15;
#pragma unroll
  for (int mi = 0; mi < 2; ++mi) {
#pragma unroll
    for (int ni = 0; ni < 4; ++ni) {
      const int gn = n0 + wn * 64 + ni * 16 + lc;
      const float bv = bias[gn];
#pragma unroll
      for (int r = 0; r < 8; ++r) {
        const int gm = m0 + wm * 32 + mi * 16 + half * 8 + r;
        const float v = acc[mi][ni][r] + bv;
        if (MODE == 1) {
          Out[(size_t)gm * N + gn] = v;
        } else {
          const int bb = gm >> 10, ss = gm & (SS - 1);
          const unsigned short hv = f32_to_bf16(v);
          if (gn < EE) {
            const int h = gn >> 7, d = gn & 127;
            Qb[(((size_t)bb * HH + h) * SS + ss) * DD + d] = hv;
          } else if (gn < 2 * EE) {
            const int e = gn - EE, h = e >> 7, d = e & 127;
            Kb[(((size_t)bb * HH + h) * SS + ss) * DD + d] = hv;
          } else {
            const int e = gn - 2 * EE, h = e >> 7, d = e & 127;
            Vtb[(((size_t)bb * HH + h) * DD + d) * SS + ss] = hv;  // V stored transposed
          }
        }
      }
    }
  }
}

// ---------------------------------------------------------------------------
// Flash attention: block = 128 queries of one (b,h); 8 waves x 16 rows each.
// KV loop over 64-key blocks. Staging uses GLOBAL_LOAD_ASYNC_TO_LDS (ASYNCcnt)
// with LDS double-buffering when available, else a register double-buffer.
// ---------------------------------------------------------------------------
__global__ __launch_bounds__(256) void flash_attn_kernel(
    const unsigned short* __restrict__ Qb, const unsigned short* __restrict__ Kb,
    const unsigned short* __restrict__ Vtb, unsigned short* __restrict__ Ctx)
{
  __shared__ unsigned short lK[2][64 * 128];   // [key][d]   2 x 16 KB
  __shared__ unsigned short lV[2][128 * 64];   // [d][key]   2 x 16 KB (from V^T)
  __shared__ unsigned short lP[8 * 16 * 64];   // per-wave P 16 KB
  const int tid = (int)threadIdx.x, lane = tid & 31, wave = tid >> 5;
  const int bh = (int)blockIdx.y;
  const int b = bh >> 4, h = bh & 15;
  const int q0 = (int)blockIdx.x * 128;
  const unsigned short* Qh = Qb + (size_t)bh * SS * DD;
  const unsigned short* Kh = Kb + (size_t)bh * SS * DD;
  const unsigned short* Vh = Vtb + (size_t)bh * DD * SS;

  // Q fragments resident in VGPRs for the whole KV loop (16 rows x 128 d)
  v16bf qf[4];
#pragma unroll
  for (int kf = 0; kf < 4; ++kf)
    qf[kf] = load_fragA(Qh + kf * 32, DD, q0 + wave * 16);

  const float cscale = 0.088388347648318447f * 1.4426950408889634f; // (1/sqrt(128))*log2(e)
  float m_i[8], l_i[8];
  v8f o[8];
  const v8f vzero = {0.f, 0.f, 0.f, 0.f, 0.f, 0.f, 0.f, 0.f};
#pragma unroll
  for (int r = 0; r < 8; ++r) { m_i[r] = -3.0e38f; l_i[r] = 0.f; }
#pragma unroll
  for (int dt = 0; dt < 8; ++dt) o[dt] = vzero;

  unsigned short* pw = lP + wave * (16 * 64);
  const int half = lane >> 4, lc = lane & 15;

  // staging geometry (per-thread, constant across the KV loop)
  const int kr = tid >> 2, kc = (tid & 3) * 32;   // K tile: [64][128]
  const int vd = tid >> 1, vc = (tid & 1) * 32;   // V tile: [128][64]
  const unsigned short* ksrc = Kh + (size_t)kr * DD + kc;
  const unsigned short* vsrc = Vh + (size_t)vd * SS + vc;
  const int NIT = SS / 64;

#if HAVE_ASYNC_LDS
  // Direct DMA into LDS, tracked by ASYNCcnt: 8x b128 per thread per tile pair.
  auto async_stage = [&](int kb, int buf) {
    const unsigned short* kg = ksrc + (size_t)kb * DD;
    const unsigned short* vg = vsrc + kb;
    unsigned short* kl = &lK[buf][kr * 128 + kc];
    unsigned short* vl = &lV[buf][vd * 64 + vc];
#pragma unroll
    for (int i = 0; i < 4; ++i)
      __builtin_amdgcn_global_load_async_to_lds_b128(
          (__attribute__((address_space(1))) i32x4*)(kg + i * 8),
          (__attribute__((address_space(3))) i32x4*)(kl + i * 8), 0, 0);
#pragma unroll
    for (int i = 0; i < 4; ++i)
      __builtin_amdgcn_global_load_async_to_lds_b128(
          (__attribute__((address_space(1))) i32x4*)(vg + i * 8),
          (__attribute__((address_space(3))) i32x4*)(vl + i * 8), 0, 0);
  };
  async_stage(0, 0);
#else
  // register double-buffer fallback
  u32x4 kreg[4], vreg[4];
#pragma unroll
  for (int i = 0; i < 4; ++i) kreg[i] = ((const u32x4*)ksrc)[i];
#pragma unroll
  for (int i = 0; i < 4; ++i) vreg[i] = ((const u32x4*)vsrc)[i];
#endif

  for (int it = 0; it < NIT; ++it) {
#if HAVE_ASYNC_LDS
    const int cur = it & 1;
    if (it + 1 < NIT) {
      async_stage((it + 1) * 64, cur ^ 1);           // DMA next buffer
      __builtin_amdgcn_s_wait_asynccnt(8);           // retire current buffer's 8 ops
    } else {
      __builtin_amdgcn_s_wait_asynccnt(0);
    }
    __syncthreads();
#else
    const int cur = 0;
    { // commit staged K/V registers to LDS
      u32x4* kd = (u32x4*)&lK[0][kr * 128 + kc];
      u32x4* vdst = (u32x4*)&lV[0][vd * 64 + vc];
#pragma unroll
      for (int i = 0; i < 4; ++i) kd[i] = kreg[i];
#pragma unroll
      for (int i = 0; i < 4; ++i) vdst[i] = vreg[i];
    }
    __syncthreads();
    if (it + 1 < NIT) {  // preload next KV block while this one is consumed
      const unsigned short* kn = ksrc + (size_t)(it + 1) * 64 * DD;
      const unsigned short* vn = vsrc + (it + 1) * 64;
#pragma unroll
      for (int i = 0; i < 4; ++i) kreg[i] = ((const u32x4*)kn)[i];
#pragma unroll
      for (int i = 0; i < 4; ++i) vreg[i] = ((const u32x4*)vn)[i];
    }
#endif
    const unsigned short* lKc = lK[cur];
    const unsigned short* lVc = lV[cur];

    // scores S = Q * K^T  (B operand: lKc is N-major [key][d])
    v8f sacc[4];
#pragma unroll
    for (int nt = 0; nt < 4; ++nt) {
      sacc[nt] = vzero;
#pragma unroll
      for (int kf = 0; kf < 4; ++kf) {
        v16bf bk = load_fragB(lKc + kf * 32, 128, nt * 16);
        sacc[nt] = wmma_bf16(qf[kf], bk, sacc[nt]);
      }
    }

    // online softmax: row max via xor-shuffle across the 16-lane half-wave
    float alpha[8];
#pragma unroll
    for (int r = 0; r < 8; ++r) {
      float rm = sacc[0][r];
#pragma unroll
      for (int nt = 1; nt < 4; ++nt) rm = fmaxf(rm, sacc[nt][r]);
#pragma unroll
      for (int mk = 1; mk < 16; mk <<= 1) rm = fmaxf(rm, __shfl_xor(rm, mk, 32));
      const float mn = fmaxf(m_i[r], rm);
      alpha[r] = exp2f((m_i[r] - mn) * cscale);
      m_i[r] = mn;
    }

    float rowsum[8];
#pragma unroll
    for (int r = 0; r < 8; ++r) rowsum[r] = 0.f;
#pragma unroll
    for (int nt = 0; nt < 4; ++nt) {
#pragma unroll
      for (int r = 0; r < 8; ++r) {
        const float p = exp2f((sacc[nt][r] - m_i[r]) * cscale);
        rowsum[r] += p;
        pw[(half * 8 + r) * 64 + nt * 16 + lc] = f32_to_bf16(p);  // C-layout -> [row][col]
      }
    }
#pragma unroll
    for (int r = 0; r < 8; ++r) {
#pragma unroll
      for (int mk = 1; mk < 16; mk <<= 1) rowsum[r] += __shfl_xor(rowsum[r], mk, 32);
      l_i[r] = l_i[r] * alpha[r] + rowsum[r];
    }

#pragma unroll
    for (int dt = 0; dt < 8; ++dt)
#pragma unroll
      for (int r = 0; r < 8; ++r) o[dt][r] = o[dt][r] * alpha[r];

    // O += P * V   (P re-read from own LDS region as A-fragments; same-wave DS is in order)
#pragma unroll
    for (int kf = 0; kf < 2; ++kf) {
      v16bf pa = load_fragA(pw + kf * 32, 64, 0);
#pragma unroll
      for (int dt = 0; dt < 8; ++dt) {
        v16bf bv = load_fragB(lVc + kf * 32, 64, dt * 16);
        o[dt] = wmma_bf16(pa, bv, o[dt]);
      }
    }
    __syncthreads();  // all DS reads of this buffer done before it is re-staged
  }

  // normalize and write ctx (bf16 [B*S][E], e = h*128 + dt*16 + col)
#pragma unroll
  for (int r = 0; r < 8; ++r) {
    const float inv = 1.0f / l_i[r];
    const int q = q0 + wave * 16 + half * 8 + r;
    unsigned short* dst = Ctx + ((size_t)b * SS + q) * EE + h * DD;
#pragma unroll
    for (int dt = 0; dt < 8; ++dt)
      dst[dt * 16 + lc] = f32_to_bf16(o[dt][r] * inv);
  }
}

// ---------------------------------------------------------------------------
extern "C" void kernel_launch(void* const* d_in, const int* in_sizes, int n_in,
                              void* d_out, int out_size, void* d_ws, size_t ws_size,
                              hipStream_t stream) {
  const float* query = (const float*)d_in[0];
  const float* in_w  = (const float*)d_in[1];   // [3E, E]
  const float* in_b  = (const float*)d_in[2];   // [3E]
  const float* out_w = (const float*)d_in[3];   // [E, E]
  const float* out_b = (const float*)d_in[4];   // [E]
  float* out = (float*)d_out;                   // [B,S,E] f32

  const size_t HEADS_ELEMS = (size_t)BB * HH * SS * DD;  // 8Mi
  unsigned short* Qb  = (unsigned short*)d_ws;
  unsigned short* Kb  = Qb + HEADS_ELEMS;
  unsigned short* Vtb = Kb + HEADS_ELEMS;
  unsigned short* Ctx = Vtb + HEADS_ELEMS;      // total 64 MB of ws

  dim3 blk(256);
  {
    dim3 g((BB * SS) / 128, (3 * EE) / 128);    // 32 x 48
    gemm_xwt_kernel<0, 0><<<g, blk, 0, stream>>>(query, in_w, in_b,
                                                 BB * SS, 3 * EE, EE,
                                                 Qb, Kb, Vtb, nullptr);
  }
  {
    dim3 g(SS / 128, BB * HH);                  // 8 x 64
    flash_attn_kernel<<<g, blk, 0, stream>>>(Qb, Kb, Vtb, Ctx);
  }
  {
    dim3 g((BB * SS) / 128, EE / 128);          // 32 x 16
    gemm_xwt_kernel<1, 1><<<g, blk, 0, stream>>>(Ctx, out_w, out_b,
                                                 BB * SS, EE, EE,
                                                 nullptr, nullptr, nullptr, out);
  }
}
